// PcdAlign_71055938945223
// MI455X (gfx1250) — matrix operand order repaired and checked by reading, not compiled, not run on hardware
//
#include <hip/hip_runtime.h>
#include <math.h>

// ---------------------------------------------------------------------------
// PCD alignment (EDVR) for MI455X / gfx1250.
//  * 3x3 conv + leaky-relu  -> implicit GEMM on v_wmma_f32_16x16x32_bf16,
//    templated on Cin so all LDS strides / tile counts are compile-time ->
//    fully unrolled constant-offset ds_load_b128 + WMMA stream.
//  * bilinear 2x upsample   -> elementwise VALU kernel
//  * deformable conv (G=3)  -> gather/VALU kernel (K=54,N=6 too small for WMMA)
// All intermediates live in d_ws (~84.4 MiB, fits easily in 192MB L2).
// ---------------------------------------------------------------------------

#define C_CH   18
#define GROUPS 3
#define CG     6
#define BATCH  4

typedef __attribute__((ext_vector_type(16))) __bf16       v16bf;
typedef __attribute__((ext_vector_type(8)))  float        v8f;
typedef __attribute__((ext_vector_type(4)))  unsigned int u32x4;
typedef __attribute__((ext_vector_type(4)))  float        f32x4;

union Frag16 { v16bf v; u32x4 q[2]; };
union Acc8   { v8f   v; f32x4 h[2]; float f[8]; };

__device__ __forceinline__ unsigned short f2bf(float f) {
  unsigned int u = __float_as_uint(f);
  u += 0x7FFFu + ((u >> 16) & 1u);          // round-to-nearest-even
  return (unsigned short)(u >> 16);
}
__device__ __forceinline__ float lrelu(float v) { return v > 0.f ? v : 0.1f * v; }
__device__ __forceinline__ int   clampi(int v, int lo, int hi) {
  return v < lo ? lo : (v > hi ? hi : v);
}

// ---------------------------------------------------------------------------
// conv3x3 (pad=1) + bias + leaky-relu, Cout=18,
// input = concat(x1[18], scale2 * x2[CIN-18]).
//
// Template parameters:
//   CIN = 18 or 36 (total input channels), RPW = output rows per wave.
// Block: 256 threads = 8 waves; tile = (8*RPW) rows x 16 cols x 18 out-ch.
// K ordering: k = tap*Cp + c (Cp = CIN padded to 32/64) so each 32-wide K tile
// sits inside a single 3x3 tap -> contiguous 16B LDS reads for the A fragment.
// B fragments (weights) are shared across the RPW rows of each wave.
// ---------------------------------------------------------------------------
template<int CIN, int RPW>
__global__ __launch_bounds__(256)
void conv3x3_lrelu_wmma(const float* __restrict__ x1,
                        const float* __restrict__ x2, float scale2,
                        const float* __restrict__ w, const float* __restrict__ bias,
                        float* __restrict__ out, int H, int W)
{
  constexpr int C1   = C_CH;
  constexpr int C2   = CIN - C_CH;
  constexpr int Cp   = (CIN <= 32) ? 32 : 64;     // padded channel count
  constexpr int KP   = 9 * Cp;                    // padded K
  constexpr int NT   = KP >> 5;                   // 32-wide K tiles (9 or 18)
  constexpr int ROWS = 8 * RPW;                   // output rows per block
  constexpr int PR   = ROWS + 2;                  // patch rows (halo)

  __shared__ unsigned short patch[PR * 18 * Cp];  // [(prow*18+pcol)*Cp + c]
  __shared__ unsigned short wlds[32 * KP];        // [n*KP + tap*Cp + c]
  __shared__ float          blds[32];

  const int tid = threadIdx.x;
  const int b   = blockIdx.z;
  const int y0  = blockIdx.y * ROWS;
  const int x0  = blockIdx.x * 16;

  // Stage weights as the WMMA B matrix (bf16, zero-padded N and C).
#pragma unroll 1
  for (int idx = tid; idx < 32 * KP; idx += 256) {
    int n   = idx / KP;
    int r   = idx - n * KP;
    int tap = r / Cp;
    int c   = r - tap * Cp;
    unsigned short v = 0;
    if (n < C_CH && c < CIN) v = f2bf(w[(n * CIN + c) * 9 + tap]);
    wlds[n * KP + tap * Cp + c] = v;
  }
  if (tid < 32) blds[tid] = (tid < C_CH) ? bias[tid] : 0.f;

  // Stage zero-padded input patch, channel-innermost (concat folded in here).
#pragma unroll 1
  for (int idx = tid; idx < PR * 18 * Cp; idx += 256) {
    int prow = idx / (18 * Cp);
    int r    = idx - prow * 18 * Cp;
    int pcol = r / Cp;
    int c    = r - pcol * Cp;
    int gy = y0 + prow - 1;
    int gx = x0 + pcol - 1;
    float v = 0.f;
    if (gy >= 0 && gy < H && gx >= 0 && gx < W && c < CIN) {
      if (c < C1) v = x1[((size_t)(b * C1 + c) * H + gy) * W + gx];
      else        v = x2[((size_t)(b * C2 + (c - C1)) * H + gy) * W + gx] * scale2;
    }
    patch[(prow * 18 + pcol) * Cp + c] = f2bf(v);
  }
  __syncthreads();

  const int lane = tid & 31;
  const int wv   = tid >> 5;          // wave id (0..7)
  const int m    = lane & 15;         // M (pixel) for A, N (channel) for B/D
  const int half = lane >> 4;

  v8f acc[RPW][2];
#pragma unroll
  for (int r = 0; r < RPW; ++r) { acc[r][0] = (v8f){}; acc[r][1] = (v8f){}; }

#pragma unroll
  for (int t = 0; t < NT; ++t) {
    const int k0  = t << 5;
    const int tap = k0 / Cp;          // entire tile lies inside one tap
    const int ky  = tap / 3, kx = tap - ky * 3;
    const int c0  = (k0 & (Cp - 1)) + half * 8;

    // B: 16-bit 32x16 layout -> K = (e<8 ? e : e+8) + half*8, N = lane%16 (two 16B chunks)
    const unsigned short* bp0 = &wlds[m * KP + k0 + half * 8];
    Frag16 B0;
    B0.q[0] = *(const u32x4*)(bp0);
    B0.q[1] = *(const u32x4*)(bp0 + 16);
    const unsigned short* bp1 = &wlds[(m + 16) * KP + k0 + half * 8];
    Frag16 B1;
    B1.q[0] = *(const u32x4*)(bp1);
    B1.q[1] = *(const u32x4*)(bp1 + 16);

#pragma unroll
    for (int r = 0; r < RPW; ++r) {
      // A: 16-bit 16x32 layout -> K = (e<8 ? e : e+8) + half*8 (two 16B chunks)
      const int prow = wv * RPW + r + ky;
      const unsigned short* ap = &patch[(prow * 18 + (m + kx)) * Cp + c0];
      Frag16 A;
      A.q[0] = *(const u32x4*)(ap);
      A.q[1] = *(const u32x4*)(ap + 16);

      acc[r][0] = __builtin_amdgcn_wmma_f32_16x16x32_bf16(
          false, A.v, false, B0.v, (short)0, acc[r][0], false, false);
      acc[r][1] = __builtin_amdgcn_wmma_f32_16x16x32_bf16(
          false, A.v, false, B1.v, (short)0, acc[r][1], false, false);
    }
  }

  // Epilogue. D layout: N = lane%16, M = r + half*8 -> two aligned b128 stores.
#pragma unroll
  for (int r = 0; r < RPW; ++r) {
    const int y = y0 + wv * RPW + r;
    {
      Acc8 o; o.v = acc[r][0];
      const float bb = blds[m];
#pragma unroll
      for (int e = 0; e < 8; ++e) o.f[e] = lrelu(o.f[e] + bb);
      float* dst = out + ((size_t)(b * C_CH + m) * H + y) * W + x0 + half * 8;
      *(f32x4*)(dst)     = o.h[0];
      *(f32x4*)(dst + 4) = o.h[1];
    }
    if (m < 2) {
      Acc8 o; o.v = acc[r][1];
      const float bb = blds[16 + m];
#pragma unroll
      for (int e = 0; e < 8; ++e) o.f[e] = lrelu(o.f[e] + bb);
      float* dst = out + ((size_t)(b * C_CH + 16 + m) * H + y) * W + x0 + half * 8;
      *(f32x4*)(dst)     = o.h[0];
      *(f32x4*)(dst + 4) = o.h[1];
    }
  }
}

// ---------------------------------------------------------------------------
// bilinear 2x upsample, align-corners scaling: s = (n_in-1)/(n_out-1)
// ---------------------------------------------------------------------------
__global__ __launch_bounds__(256)
void up2_kernel(const float* __restrict__ in, float* __restrict__ out,
                int BC, int Hi, int Wi)
{
  const int Ho = Hi * 2, Wo = Wi * 2;
  const size_t total = (size_t)BC * Ho * Wo;
  size_t i = (size_t)blockIdx.x * 256 + threadIdx.x;
  if (i >= total) return;
  int xo = (int)(i % Wo);
  size_t r = i / Wo;
  int yo = (int)(r % Ho);
  int bc = (int)(r / Ho);
  float sy = yo * (float)(Hi - 1) / (float)(Ho - 1);
  float sx = xo * (float)(Wi - 1) / (float)(Wo - 1);
  int ya = (int)floorf(sy), xa = (int)floorf(sx);
  int yb = ya + 1 < Hi ? ya + 1 : Hi - 1;
  int xb = xa + 1 < Wi ? xa + 1 : Wi - 1;
  float wy = sy - (float)ya, wx = sx - (float)xa;
  const float* p = in + (size_t)bc * Hi * Wi;
  float v00 = p[(size_t)ya * Wi + xa], v01 = p[(size_t)ya * Wi + xb];
  float v10 = p[(size_t)yb * Wi + xa], v11 = p[(size_t)yb * Wi + xb];
  out[i] = (v00 * (1.f - wx) + v01 * wx) * (1.f - wy)
         + (v10 * (1.f - wx) + v11 * wx) * wy;
}

// ---------------------------------------------------------------------------
// deformable conv 3x3, groups=3, Cg=Cog=6, + bias (+ optional leaky-relu)
// One thread per (b,g,pixel); per-group weight slice staged in LDS.
// ---------------------------------------------------------------------------
__global__ __launch_bounds__(256)
void dcn_kernel(const float* __restrict__ x, const float* __restrict__ off,
                const float* __restrict__ w, const float* __restrict__ bias,
                float* __restrict__ out, int H, int W, int do_lrelu)
{
  __shared__ float wl[CG][CG][9];    // [o][c][k]
  __shared__ float bl[CG];
  const int tid = threadIdx.x;
  const int bg  = blockIdx.y;
  const int b   = bg / GROUPS, g = bg % GROUPS;

  for (int idx = tid; idx < CG * CG * 9; idx += 256) {
    int o = idx / (CG * 9);
    int r = idx - o * CG * 9;
    int c = r / 9, k = r - c * 9;
    wl[o][c][k] = w[((size_t)(g * CG + o) * CG + c) * 9 + k];
  }
  if (tid < CG) bl[tid] = bias[g * CG + tid];
  __syncthreads();

  const int pix = blockIdx.x * 256 + tid;
  if (pix >= H * W) return;
  const int y = pix / W, xq = pix - y * W;
  const size_t HW = (size_t)H * W;

  float acc[CG];
#pragma unroll
  for (int o = 0; o < CG; ++o) acc[o] = 0.f;

  const float* offb = off + (size_t)b * C_CH * HW + pix;
  const float* base = x + (size_t)(b * C_CH + g * CG) * HW;

#pragma unroll
  for (int k = 0; k < 9; ++k) {
    const float dy = offb[(size_t)(2 * k) * HW];
    const float dx = offb[(size_t)(2 * k + 1) * HW];
    const int ky = k / 3, kx = k - ky * 3;
    const float py = (float)(y + ky - 1) + dy;
    const float px = (float)(xq + kx - 1) + dx;
    const float yf = floorf(py), xf = floorf(px);
    const int ya = (int)yf, xa = (int)xf;
    const int yb = ya + 1, xb = xa + 1;
    const float wy = py - yf, wx = px - xf;
    const float vy0 = (ya >= 0 && ya < H) ? 1.f : 0.f;
    const float vy1 = (yb >= 0 && yb < H) ? 1.f : 0.f;
    const float vx0 = (xa >= 0 && xa < W) ? 1.f : 0.f;
    const float vx1 = (xb >= 0 && xb < W) ? 1.f : 0.f;
    const int yc0 = clampi(ya, 0, H - 1), yc1 = clampi(yb, 0, H - 1);
    const int xc0 = clampi(xa, 0, W - 1), xc1 = clampi(xb, 0, W - 1);
    const float w00 = (1.f - wy) * (1.f - wx) * vy0 * vx0;
    const float w01 = (1.f - wy) * wx * vy0 * vx1;
    const float w10 = wy * (1.f - wx) * vy1 * vx0;
    const float w11 = wy * wx * vy1 * vx1;
#pragma unroll
    for (int c = 0; c < CG; ++c) {
      const float* pc = base + (size_t)c * HW;
      const float v = pc[(size_t)yc0 * W + xc0] * w00
                    + pc[(size_t)yc0 * W + xc1] * w01
                    + pc[(size_t)yc1 * W + xc0] * w10
                    + pc[(size_t)yc1 * W + xc1] * w11;
#pragma unroll
      for (int o = 0; o < CG; ++o) acc[o] += v * wl[o][c][k];
    }
  }

  float* ob = out + (size_t)(b * C_CH + g * CG) * HW + pix;
#pragma unroll
  for (int o = 0; o < CG; ++o) {
    float v = acc[o] + bl[o];
    ob[(size_t)o * HW] = do_lrelu ? lrelu(v) : v;
  }
}

// ---------------------------------------------------------------------------
// Host-side orchestration
// ---------------------------------------------------------------------------
static void conv_l(const float* x1, const float* x2, int C2, float s2,
                   const float* w, const float* b, float* out,
                   int H, int W, hipStream_t st)
{
  if (C2 == 0) {
    // Cin=18: 2 rows/wave, 16-row tiles (B fragments reused across rows)
    dim3 grid(W / 16, H / 16, BATCH);
    conv3x3_lrelu_wmma<18, 2><<<grid, 256, 0, st>>>(x1, nullptr, 1.f, w, b, out, H, W);
  } else {
    // Cin=36: 1 row/wave, 8-row tiles (keeps LDS ~60KB)
    dim3 grid(W / 16, H / 8, BATCH);
    conv3x3_lrelu_wmma<36, 1><<<grid, 256, 0, st>>>(x1, x2, s2, w, b, out, H, W);
  }
}

static void up2_l(const float* in, float* out, int Hi, int Wi, hipStream_t st)
{
  const size_t total = (size_t)BATCH * C_CH * (2 * Hi) * (2 * Wi);
  up2_kernel<<<(unsigned)((total + 255) / 256), 256, 0, st>>>(
      in, out, BATCH * C_CH, Hi, Wi);
}

static void dcn_l(const float* x, const float* off, const float* w,
                  const float* b, float* out, int H, int W, int do_lrelu,
                  hipStream_t st)
{
  dim3 grid((H * W + 255) / 256, BATCH * GROUPS);
  dcn_kernel<<<grid, 256, 0, st>>>(x, off, w, b, out, H, W, do_lrelu);
}

extern "C" void kernel_launch(void* const* d_in, const int* in_sizes, int n_in,
                              void* d_out, int out_size, void* d_ws, size_t ws_size,
                              hipStream_t stream)
{
  const float* nbr1 = (const float*)d_in[0];
  const float* nbr2 = (const float*)d_in[1];
  const float* nbr3 = (const float*)d_in[2];
  const float* ref1 = (const float*)d_in[3];
  const float* ref2 = (const float*)d_in[4];
  const float* ref3 = (const float*)d_in[5];
  const float* const* P = (const float* const*)(d_in + 6);  // 32 params in spec order

  const float *s3_c1_w = P[0],  *s3_c1_b = P[1],  *s3_c2_w = P[2],  *s3_c2_b = P[3];
  const float *s3_dcn_w = P[4], *s3_dcn_b = P[5];
  const float *s2_off1_w = P[6],  *s2_off1_b = P[7];
  const float *s2_off2_w = P[8],  *s2_off2_b = P[9];
  const float *s2_off3_w = P[10], *s2_off3_b = P[11];
  const float *s2_dcn_w = P[12],  *s2_dcn_b = P[13];
  const float *s2_fea_w = P[14],  *s2_fea_b = P[15];
  const float *s1_off1_w = P[16], *s1_off1_b = P[17];
  const float *s1_off2_w = P[18], *s1_off2_b = P[19];
  const float *s1_off3_w = P[20], *s1_off3_b = P[21];
  const float *s1_dcn_w = P[22],  *s1_dcn_b = P[23];
  const float *s1_fea_w = P[24],  *s1_fea_b = P[25];
  const float *cas_off1_w = P[26], *cas_off1_b = P[27];
  const float *cas_off2_w = P[28], *cas_off2_b = P[29];
  const float *cas_dcn_w = P[30],  *cas_dcn_b = P[31];

  const int H1 = 256, W1 = 256, H2 = 128, W2 = 128, H3 = 64, W3 = 64;
  const size_t Q  = (size_t)BATCH * C_CH * H3 * W3;   //  294912 floats
  const size_t Hs = (size_t)BATCH * C_CH * H2 * W2;   // 1179648 floats
  // workspace carve-up (3 quarter + 6 half + 3 full ~= 84.4 MiB)
  float* q1 = (float*)d_ws;
  float* q2 = q1 + Q;
  float* q3 = q2 + Q;
  float* h1 = q3 + Q;
  float* h2 = h1 + Hs;
  float* h3 = h2 + Hs;
  float* h4 = h3 + Hs;
  float* h5 = h4 + Hs;
  float* h6 = h5 + Hs;
  float* fb1 = h6 + Hs;
  float* fb2 = fb1 + (size_t)BATCH * C_CH * H1 * W1;
  float* fb3 = fb2 + (size_t)BATCH * C_CH * H1 * W1;
  float* outp = (float*)d_out;

  // ---- Level 3 (64x64) ----
  conv_l(nbr3, ref3, C_CH, 1.f, s3_c1_w, s3_c1_b, q1, H3, W3, stream);
  conv_l(q1, nullptr, 0, 1.f, s3_c2_w, s3_c2_b, q2, H3, W3, stream);         // o3
  dcn_l(nbr3, q2, s3_dcn_w, s3_dcn_b, q3, H3, W3, /*lrelu=*/1, stream);      // f3
  up2_l(q2, h1, H3, W3, stream);                                             // up(o3)

  // ---- Level 2 (128x128) ----
  conv_l(nbr2, ref2, C_CH, 1.f, s2_off1_w, s2_off1_b, h2, H2, W2, stream);
  conv_l(h2, h1, C_CH, 2.f, s2_off2_w, s2_off2_b, h3, H2, W2, stream);       // *2 offset
  conv_l(h3, nullptr, 0, 1.f, s2_off3_w, s2_off3_b, h4, H2, W2, stream);     // o2
  dcn_l(nbr2, h4, s2_dcn_w, s2_dcn_b, h5, H2, W2, /*lrelu=*/0, stream);      // f
  up2_l(q3, h6, H3, W3, stream);                                             // up(f3)
  conv_l(h5, h6, C_CH, 1.f, s2_fea_w, s2_fea_b, h2, H2, W2, stream);         // f2
  up2_l(h4, fb1, H2, W2, stream);                                            // up(o2)

  // ---- Level 1 (256x256) ----
  conv_l(nbr1, ref1, C_CH, 1.f, s1_off1_w, s1_off1_b, fb2, H1, W1, stream);
  conv_l(fb2, fb1, C_CH, 2.f, s1_off2_w, s1_off2_b, fb3, H1, W1, stream);
  conv_l(fb3, nullptr, 0, 1.f, s1_off3_w, s1_off3_b, fb1, H1, W1, stream);   // o1
  dcn_l(nbr1, fb1, s1_dcn_w, s1_dcn_b, fb2, H1, W1, /*lrelu=*/0, stream);    // f
  up2_l(h2, fb3, H2, W2, stream);                                            // up(f2)
  conv_l(fb2, fb3, C_CH, 1.f, s1_fea_w, s1_fea_b, fb1, H1, W1, stream);      // f1

  // ---- Cascade ----
  conv_l(fb1, ref1, C_CH, 1.f, cas_off1_w, cas_off1_b, fb2, H1, W1, stream);
  conv_l(fb2, nullptr, 0, 1.f, cas_off2_w, cas_off2_b, fb3, H1, W1, stream); // co
  dcn_l(fb1, fb3, cas_dcn_w, cas_dcn_b, outp, H1, W1, /*lrelu=*/1, stream);
}
